// fGCNModelVAE_23562190586108
// MI455X (gfx1250) — matrix operand order repaired and checked by reading, not compiled, not run on hardware
//
#include <hip/hip_runtime.h>
#include <hip/hip_bf16.h>

// ---------------------------------------------------------------------------
// GCN-VAE forward for MI455X (gfx1250, wave32, WMMA).
// All GEMM operands pre-converted to bf16 (K-stride padded to %32==0), weights
// pre-transposed -> one unified WMMA GEMM kernel. Ping-pong double-buffered
// LDS tiles with rotating pointers: global loads for step k+1 issue before the
// 8 WMMAs of step k and are stored to the inactive buffer in the SAME
// iteration (no register lives across a barrier, single compute call site).
// Graph scatter uses hw f32 atomics (L2-resident).
// pred = mu@mu^T (400MB) streams out via nontemporal stores.
// Tiling: block 128x128, BK=32, 8 waves, wave tile 32x64 (8 WMMA / k-step).
// LDS rows padded to 40 shorts -> conflict-free ds_load_b128 fragments.
// ---------------------------------------------------------------------------

typedef __attribute__((ext_vector_type(16))) __bf16 v16bf;
typedef __attribute__((ext_vector_type(8)))  float  v8f;

union Frag16 { v16bf v; uint4 q[2]; };

constexpr int LDST = 40;  // LDS row stride (shorts): 80B, 16B-aligned, conflict-free

__device__ __forceinline__ unsigned short f2bf(float f) {
    unsigned int u = __float_as_uint(f);
    return (unsigned short)((u + 0x8000u) >> 16);  // round-half-up
}

// pack bf16(lo) | bf16(hi)<<16 : two v_add + one v_perm_b32
__device__ __forceinline__ unsigned int pk_bf16(float lo, float hi) {
    unsigned int a = __float_as_uint(lo) + 0x8000u;
    unsigned int b = __float_as_uint(hi) + 0x8000u;
    return __builtin_amdgcn_perm(b, a, 0x07060302u);
}

// A-fragment (16x32 bf16) from LDS row-major tile, row stride LDST shorts.
// lane<16: row=lane, K={0..7,16..23}; lane>=16: K={8..15,24..31}.
__device__ __forceinline__ Frag16 load_fragA(const unsigned short* tile, int row0,
                                             int lane) {
    int half = lane >> 4, r = lane & 15;
    Frag16 f;
    f.q[0] = *(const uint4*)&tile[(row0 + r) * LDST + half * 8];
    f.q[1] = *(const uint4*)&tile[(row0 + r) * LDST + 16 + half * 8];
    return f;
}

// B-fragment (32x16 bf16) from LDS tile stored [n][k], row stride LDST shorts.
// lanes 0-15: K={0..7,16..23}, lanes 16-31: K={8..15,24..31}, col n = lane&15.
__device__ __forceinline__ Frag16 load_fragB(const unsigned short* tile, int col0,
                                             int lane) {
    int half = lane >> 4, c = lane & 15;
    Frag16 f;
    f.q[0] = *(const uint4*)&tile[(col0 + c) * LDST + half * 8];
    f.q[1] = *(const uint4*)&tile[(col0 + c) * LDST + 16 + half * 8];
    return f;
}

// ---------------------------------------------------------------------------
// Unified bf16 GEMM: C[M,N] = A[M,K] @ Bt[N,K]^T. lda/ldb in shorts, K%32==0,
// A rows / Bt rows clamped (out-of-range rows feed never-stored C entries).
// NT: nontemporal C stores (streaming outputs).
// ---------------------------------------------------------------------------
template <bool NT>
__global__ __launch_bounds__(256, 1) void gemm_bb_wmma(
    const unsigned short* __restrict__ A, int lda,
    const unsigned short* __restrict__ Bt, int ldb,
    float* __restrict__ C, int ldc, int M, int N, int K) {
    __shared__ unsigned short As0[128 * LDST];
    __shared__ unsigned short Bs0[128 * LDST];
    __shared__ unsigned short As1[128 * LDST];
    __shared__ unsigned short Bs1[128 * LDST];

    const int t    = threadIdx.x;
    const int bm0  = blockIdx.x * 128;
    const int bn0  = blockIdx.y * 128;
    const int lane = t & 31;
    const int wv   = t >> 5;
    const int wm   = (wv >> 1) * 32;   // 0,32,64,96
    const int wn   = (wv & 1) * 64;    // 0,64
    const int half = lane >> 4, l15 = lane & 15;

    // 2 slots/thread over 128 rows x 4 quads-of-8-shorts (16B each)
    int rowL[2], qL[2];
    const unsigned short* aPtr[2];
    const unsigned short* bPtr[2];
#pragma unroll
    for (int i = 0; i < 2; ++i) {
        int s = t + i * 256;
        int r = s >> 2, q = (s & 3) * 8;
        rowL[i] = r; qL[i] = q;
        int ga = bm0 + r; if (ga > M - 1) ga = M - 1;  // clamp: never stored
        int gb = bn0 + r; if (gb > N - 1) gb = N - 1;
        aPtr[i] = A  + (size_t)ga * lda + q;
        bPtr[i] = Bt + (size_t)gb * ldb + q;
    }

    uint4 aReg[2], bReg[2];
    v8f acc[2][4] = {};

    auto loadTiles = [&](int k0) {
#pragma unroll
        for (int i = 0; i < 2; ++i) {
            aReg[i] = *(const uint4*)(aPtr[i] + k0);
            bReg[i] = *(const uint4*)(bPtr[i] + k0);
        }
    };
    auto storeTiles = [&](unsigned short* as, unsigned short* bs) {
#pragma unroll
        for (int i = 0; i < 2; ++i) {
            *(uint4*)&as[rowL[i] * LDST + qL[i]] = aReg[i];
            *(uint4*)&bs[rowL[i] * LDST + qL[i]] = bReg[i];
        }
    };

    const int nk = K >> 5;  // K % 32 == 0 (operands pre-padded)
    loadTiles(0);
    storeTiles(As0, Bs0);

    const unsigned short* cA = As0;  // compute buffer
    const unsigned short* cB = Bs0;
    unsigned short* sA = As1;        // staging buffer
    unsigned short* sB = Bs1;

    for (int kk = 0; kk < nk; ++kk) {
        __syncthreads();  // staging of cur buffer (prev iter) now visible
        bool more = (kk + 1) < nk;
        if (more) loadTiles((kk + 1) * 32);  // loads fly over the WMMAs
        {
            Frag16 a[2], b[4];
#pragma unroll
            for (int i = 0; i < 2; ++i) a[i] = load_fragA(cA, wm + 16 * i, lane);
#pragma unroll
            for (int j = 0; j < 4; ++j) b[j] = load_fragB(cB, wn + 16 * j, lane);
#pragma unroll
            for (int i = 0; i < 2; ++i)
#pragma unroll
                for (int j = 0; j < 4; ++j)
                    acc[i][j] = __builtin_amdgcn_wmma_f32_16x16x32_bf16(
                        false, a[i].v, false, b[j].v, (short)0, acc[i][j], false, false);
        }
        if (more) storeTiles(sA, sB);  // safe: sA/sB reads finished pre-barrier
        // ping-pong
        const unsigned short* tA = cA;
        const unsigned short* tB = cB;
        cA = sA; cB = sB;
        sA = (unsigned short*)tA; sB = (unsigned short*)tB;
    }

    // fragment elem e -> M = 8*half + e, N = lane&15
#pragma unroll
    for (int i = 0; i < 2; ++i)
#pragma unroll
        for (int j = 0; j < 4; ++j)
#pragma unroll
            for (int e = 0; e < 8; ++e) {
                int gr = bm0 + wm + 16 * i + 8 * half + e;
                int gc = bn0 + wn + 16 * j + l15;
                if (gr < M && gc < N) {
                    float* p = &C[(size_t)gr * ldc + gc];
                    if (NT) __builtin_nontemporal_store(acc[i][j][e], p);
                    else    *p = acc[i][j][e];
                }
            }
}

// ---------------------------------------------------------------------------
// f32 [R][ldi] -> bf16 [R][ldo], zero-pad cols K..ldo-1, optional fused relu.
// ---------------------------------------------------------------------------
template <bool RELU>
__global__ void cvt_rows(const float* __restrict__ in, int K, int ldi,
                         unsigned short* __restrict__ out, int ldo, int R) {
    int i = blockIdx.x * blockDim.x + threadIdx.x;
    int perRow = ldo >> 2;
    int r = i / perRow;
    if (r >= R) return;
    int c4 = (i - r * perRow) * 4;
    float x[4] = {0.f, 0.f, 0.f, 0.f};
    if (c4 + 3 < K) {
        float4 f = *(const float4*)&in[(size_t)r * ldi + c4];
        x[0] = f.x; x[1] = f.y; x[2] = f.z; x[3] = f.w;
    } else {
#pragma unroll
        for (int j = 0; j < 4; ++j)
            if (c4 + j < K) x[j] = in[(size_t)r * ldi + c4 + j];
    }
    if (RELU) {
#pragma unroll
        for (int j = 0; j < 4; ++j) x[j] = fmaxf(x[j], 0.f);
    }
    uint2 u;
    u.x = pk_bf16(x[0], x[1]);
    u.y = pk_bf16(x[2], x[3]);
    *(uint2*)&out[(size_t)r * ldo + c4] = u;
}

// ---------------------------------------------------------------------------
// Transpose-convert: out[n][k] = bf16(in[k][n]), zero-pad k in [K, ldo).
// (one-time weight prep; reads coalesced along n)
// ---------------------------------------------------------------------------
__global__ void cvt_transpose(const float* __restrict__ in, int K, int Ncols, int ldi,
                              unsigned short* __restrict__ out, int ldo) {
    int i = blockIdx.x * blockDim.x + threadIdx.x;
    int k = i / Ncols;
    if (k >= ldo) return;
    int n = i - k * Ncols;
    float v = (k < K) ? in[(size_t)k * ldi + n] : 0.f;
    out[(size_t)n * ldo + k] = f2bf(v);
}

// ---------------------------------------------------------------------------
// Edge scatter-add: out[dst[e]][c] += w[e] * S[src[e]][c]; blockDim == H.
// ---------------------------------------------------------------------------
__global__ void scatter_edges(const float* __restrict__ S, const int* __restrict__ src,
                              const int* __restrict__ dst, const float* __restrict__ w,
                              float* __restrict__ out, int E, int H) {
    int e = blockIdx.x;
    if (e >= E) return;
    int c = threadIdx.x;
    int s = src[e], d = dst[e];
    float v = w[e] * S[(size_t)s * H + c];
    unsafeAtomicAdd(&out[(size_t)d * H + c], v);
}

__global__ void zero_f32(float* __restrict__ p, int n) {
    int i = blockIdx.x * blockDim.x + threadIdx.x;
    if (i < n) p[i] = 0.0f;
}

__global__ void split_mu_logvar(const float* __restrict__ out23,
                                float* __restrict__ mu, float* __restrict__ lv, int Nn) {
    int i = blockIdx.x * blockDim.x + threadIdx.x;
    if (i >= Nn * 64) return;
    int r = i >> 6, c = i & 63;
    mu[i] = out23[(size_t)r * 128 + c];
    lv[i] = out23[(size_t)r * 128 + 64 + c];
}

extern "C" void kernel_launch(void* const* d_in, const int* in_sizes, int n_in,
                              void* d_out, int out_size, void* d_ws, size_t ws_size,
                              hipStream_t stream) {
    const float* x    = (const float*)d_in[0];
    const int*   esrc = (const int*)d_in[1];
    const int*   edst = (const int*)d_in[2];
    const float* ew   = (const float*)d_in[3];
    const float* W1   = (const float*)d_in[4];
    const float* W2   = (const float*)d_in[5];
    const float* W3   = (const float*)d_in[6];

    const int Nn = 10000, Fin = 3000, FinP = 3008;  // FinP: K padded to %32
    const int H1 = 512, H2 = 64, H23 = 128;
    const int E = in_sizes[1];

    // ---- workspace layout (~104.2 MB), lifetime-safe overlaying ----
    char* w = (char*)d_ws;
    unsigned short* xb   = (unsigned short*)(w);                    // 10000*3008*2 = 60,160,000
    unsigned short* W1tb = (unsigned short*)(w + 60160000);         // 512*3008*2  =  3,080,192
    float* support1      = (float*)(w + 63240192);                  // 10000*512*4 = 20,480,000
    float* out1          = (float*)(w + 83720192);                  // 10000*512*4 = 20,480,000
    // overlay inside xb region (xb dead after gemm1):
    unsigned short* h1b  = (unsigned short*)(w);                    // 10000*512*2 = 10,240,000
    unsigned short* Wt23 = (unsigned short*)(w + 10240000);         // 128*512*2   =    131,072
    float* support23     = (float*)(w + 10371072);                  // 10000*128*4 =  5,120,000
    float* out23         = (float*)(w + 15491072);                  // 10000*128*4 =  5,120,000
    unsigned short* Zb   = (unsigned short*)(w + 20611072);         // 10000*64*2  =  1,280,000

    float* pred = (float*)d_out;               // Nn*Nn
    float* mu   = pred + (size_t)Nn * Nn;      // Nn*H2
    float* lv   = mu + (size_t)Nn * H2;        // Nn*H2

    // out1 zeroed up front (region not overlaid)
    zero_f32<<<(Nn * H1 + 255) / 256, 256, 0, stream>>>(out1, Nn * H1);

    // ---- operand prep for layer 1 ----
    {
        int tot = Nn * (FinP / 4);
        cvt_rows<false><<<(tot + 255) / 256, 256, 0, stream>>>(x, Fin, Fin, xb, FinP, Nn);
    }
    {
        int tot = FinP * H1;
        cvt_transpose<<<(tot + 255) / 256, 256, 0, stream>>>(W1, Fin, H1, H1, W1tb, FinP);
    }

    // Layer 1: support1 = x @ W1
    dim3 g1((Nn + 127) / 128, H1 / 128);
    gemm_bb_wmma<false><<<g1, 256, 0, stream>>>(xb, FinP, W1tb, FinP,
                                                support1, H1, Nn, H1, FinP);
    scatter_edges<<<E, H1, 0, stream>>>(support1, esrc, edst, ew, out1, E, H1);

    // ---- operand prep for layers 2+3 (xb region now dead -> overlay) ----
    {
        int tot = Nn * (H1 / 4);
        cvt_rows<true><<<(tot + 255) / 256, 256, 0, stream>>>(out1, H1, H1, h1b, H1, Nn);
    }
    {
        int tot = H1 * H2;
        cvt_transpose<<<(tot + 255) / 256, 256, 0, stream>>>(W2, H1, H2, H2, Wt23, H1);
        cvt_transpose<<<(tot + 255) / 256, 256, 0, stream>>>(W3, H1, H2, H2,
                                                             Wt23 + (size_t)H2 * H1, H1);
    }

    // Layers 2+3 fused: support23 = relu(out1) @ [W2 | W3]
    dim3 g2((Nn + 127) / 128, 1);
    gemm_bb_wmma<false><<<g2, 256, 0, stream>>>(h1b, H1, Wt23, H1,
                                                support23, H23, Nn, H23, H1);
    zero_f32<<<(Nn * H23 + 255) / 256, 256, 0, stream>>>(out23, Nn * H23);
    scatter_edges<<<E, H23, 0, stream>>>(support23, esrc, edst, ew, out23, E, H23);

    split_mu_logvar<<<(Nn * H2 + 255) / 256, 256, 0, stream>>>(out23, mu, lv, Nn);

    // Decoder: pred = mu @ mu^T (Zb is both A and Bt), streaming NT stores
    {
        int tot = Nn * (H2 / 4);
        cvt_rows<false><<<(tot + 255) / 256, 256, 0, stream>>>(out23, H2, H23, Zb, H2, Nn);
    }
    dim3 g3((Nn + 127) / 128, (Nn + 127) / 128);
    gemm_bb_wmma<true><<<g3, 256, 0, stream>>>(Zb, H2, Zb, H2,
                                               pred, Nn, Nn, Nn, H2);
}
